// SimplePerformerSeqModel_33243046871137
// MI455X (gfx1250) — matrix operand order repaired and checked by reading, not compiled, not run on hardware
//
#include <hip/hip_runtime.h>
#include <hip/hip_bf16.h>
#include <math.h>

// ---------------- problem constants (match reference) ----------------
#define BATCH    2
#define LSEQ     4096
#define DMODEL   128
#define PROJ     64
#define NLAYERS  2
#define VOCAB    32000
#define MTOK     (BATCH * LSEQ)        // 8192 rows
#define DFF      (4 * DMODEL)          // 512

typedef __attribute__((ext_vector_type(16))) _Float16 v16h;
typedef __attribute__((ext_vector_type(8)))  _Float16 v8h;
typedef __attribute__((ext_vector_type(4)))  _Float16 v4h;
typedef __attribute__((ext_vector_type(2)))  _Float16 v2h;
typedef __attribute__((ext_vector_type(8)))  float    v8f;
typedef __attribute__((ext_vector_type(4)))  unsigned int u32x4;
typedef __attribute__((ext_vector_type(8)))  int      i32x8;
typedef __attribute__((ext_vector_type(4)))  int      i32x4;

// LDS stride in halves: 48 => 96 bytes/row, keeps v16h loads 32B-aligned, pads banks.
// TDM padding (16 DWORDs stored -> skip 8 DWORDs) reproduces this stride in the DMA.
#define LDSTR 48

#if __has_builtin(__builtin_amdgcn_tensor_load_to_lds) && __has_builtin(__builtin_amdgcn_s_wait_tensorcnt)
#define USE_TDM 1
#else
#define USE_TDM 0
#endif

#if USE_TDM
// ---------------------------------------------------------------------
// Issue one TDM descriptor: 2-D tile of f16, rows x cols, row stride
// `stride_elems`, into LDS at byte offset lds_off with an 8-DWORD pad
// after every 16 DWORDs (=> LDS row stride 48 halves for 32-half rows).
// Wave-uniform; call from a single wave. 6-arg builtin (clang-23 lane).
// ---------------------------------------------------------------------
__device__ __forceinline__ void tdm_load_tile_f16(const _Float16* gptr, unsigned int lds_off,
                                                  unsigned int cols, unsigned int rows,
                                                  unsigned long long stride_elems)
{
    unsigned long long ga = (unsigned long long)(uintptr_t)gptr;
    u32x4 g0;
    g0[0] = 1u;                                        // count=1, user descriptor
    g0[1] = lds_off;                                   // LDS byte address
    g0[2] = (unsigned int)(ga & 0xFFFFFFFFu);          // global_addr[31:0]
    g0[3] = (unsigned int)((ga >> 32) & 0x1FFFFFFu)    // global_addr[56:32]
          | (2u << 30);                                // type = 2 ("image")
    i32x8 g1;
    unsigned int d0 = (1u << 16)                       // data_size = 2 bytes
                    | (1u << 20)                       // pad_enable
                    | (3u << 22)                       // pad_interval: 16 DWORDs
                    | (7u << 25);                      // pad_amount:   8 DWORDs
    g1[0] = (int)d0;
    g1[1] = (int)((cols & 0xFFFFu) << 16);                                 // tensor_dim0[15:0]
    g1[2] = (int)(((cols >> 16) & 0xFFFFu) | ((rows & 0xFFFFu) << 16));    // dim0 hi | dim1 lo
    g1[3] = (int)(((rows >> 16) & 0xFFFFu) | ((cols & 0xFFFFu) << 16));    // dim1 hi | tile_dim0
    g1[4] = (int)(rows & 0xFFFFu);                                         // tile_dim1 | tile_dim2=0
    g1[5] = (int)(unsigned int)(stride_elems & 0xFFFFFFFFull);             // dim0_stride lo
    g1[6] = (int)(unsigned int)((stride_elems >> 32) & 0xFFFFull);         // dim0_stride hi | dim1_stride lo
    g1[7] = 0;
    i32x4 z4 = {0, 0, 0, 0};
    i32x8 z8 = {0, 0, 0, 0, 0, 0, 0, 0};
    __builtin_amdgcn_tensor_load_to_lds(g0, g1, z4, z4, z8, 0);
}
#endif

// ---------------------------------------------------------------------
// Generic f16 WMMA GEMM:  C[M,N] = op(A[M,K] @ B[K,N] + bias) (*rowscale) (+res)
// Block tile: 128(M) x 64(N), K-step 32. 8 waves: 4(M) x 2(N), wave tile 32x32.
// Double-buffered software pipeline: while WMMAs consume buffer `cur`,
// the Tensor Data Mover (wave 0) DMAs the next A-tile and all waves stage
// the next (transposed) B-tile into buffer `nxt`. s_wait_tensorcnt +
// barrier land after the compute, overlapping DMA with math.
// OP: 0 = none, 1 = exact GELU, 2 = elu(x)+1
// ---------------------------------------------------------------------
template<int OP, bool BIAS, bool STORE_F32, bool STORE_F16, bool ADD_RES, bool ROWSCALE>
__global__ __launch_bounds__(256)
void gemm_f16_kernel(const _Float16* __restrict__ A, const _Float16* __restrict__ B,
                     const float* __restrict__ bias, const float* __restrict__ res,
                     const float* __restrict__ rowscale,
                     float* __restrict__ Cf, _Float16* __restrict__ Ch,
                     int M, int N, int K)
{
    __shared__ _Float16 As[2][128 * LDSTR];   // A tiles, row-major [row][k], stride 48
    __shared__ _Float16 Bs[2][64 * LDSTR];    // B tiles, K-transposed [n][k], stride 48

    const int t     = threadIdx.x;
    const int lane  = t & 31;
    const int w     = t >> 5;      // 0..7
    const int waveM = w & 3;       // 0..3
    const int waveN = w >> 2;      // 0..1
    const int bm    = blockIdx.y * 128;
    const int bn    = blockIdx.x * 64;

    v8f acc[2][2] = {};

    const int r  = lane & 15;
    const int hk = lane >> 4;

    const _Float16* Ablk = A + (size_t)bm * K;

    // ---- staging helpers ----
    auto stageA = [&](int k0, int buf) {
#if USE_TDM
        if (w == 0) {
            unsigned int lds_off = (unsigned int)(uintptr_t)(&As[buf][0]);
            tdm_load_tile_f16(Ablk + k0, lds_off, 32u, 128u, (unsigned long long)K);
        }
#else
        for (int c = t; c < 512; c += 256) {
            int row = c >> 2;
            int col = (c & 3) * 8;
            v8h tmp = *(const v8h*)(Ablk + (size_t)row * K + k0 + col);
            *(v8h*)(&As[buf][0] + row * LDSTR + col) = tmp;
        }
#endif
    };
    auto stageB = [&](int k0, int buf) {
        int kp = t >> 4;       // 0..15 -> k = 2*kp
        int g  = t & 15;       // n-group of 4
        const _Float16* bp = B + (size_t)(k0 + 2 * kp) * N + bn + g * 4;
        v4h r0 = *(const v4h*)bp;
        v4h r1 = *(const v4h*)(bp + N);
        #pragma unroll
        for (int j = 0; j < 4; ++j) {
            v2h pr; pr[0] = r0[j]; pr[1] = r1[j];
            *(v2h*)(&Bs[buf][0] + (g * 4 + j) * LDSTR + 2 * kp) = pr;
        }
    };

    // ---- prologue: fill buffer 0 ----
    stageA(0, 0);
    stageB(0, 0);
#if USE_TDM
    if (w == 0) __builtin_amdgcn_s_wait_tensorcnt(0);
#endif
    __syncthreads();

    for (int k0 = 0; k0 < K; k0 += 32) {
        const int cur = (k0 >> 5) & 1;
        const int nxt = cur ^ 1;
        const bool more = (k0 + 32) < K;

        // ---- stage next tiles into the other buffer (overlaps with compute) ----
        if (more) {
            stageA(k0 + 32, nxt);
            stageB(k0 + 32, nxt);
        }

        // ---- B fragments for this wave's two 16-col tiles ----
        const _Float16* Bsc = &Bs[cur][0];
        const _Float16* Asc = &As[cur][0];
        v16h bfrag[2];
        #pragma unroll
        for (int ni = 0; ni < 2; ++ni) {
            int c = waveN * 32 + ni * 16 + r;
            bfrag[ni] = *(const v16h*)(Bsc + c * LDSTR + hk * 16);
        }
        // ---- A fragments + WMMA ----
        #pragma unroll
        for (int mi = 0; mi < 2; ++mi) {
            int row = waveM * 32 + mi * 16 + r;
            v8h lo = *(const v8h*)(Asc + row * LDSTR + hk * 16);
            v8h hi = *(const v8h*)(Asc + row * LDSTR + hk * 16 + 8);
            v16h afrag;
            #pragma unroll
            for (int i = 0; i < 8; ++i) { afrag[i] = lo[i]; afrag[8 + i] = hi[i]; }
            #pragma unroll
            for (int ni = 0; ni < 2; ++ni)
                acc[mi][ni] = __builtin_amdgcn_wmma_f32_16x16x32_f16(
                    false, afrag, false, bfrag[ni], (short)0, acc[mi][ni], false, false);
        }

#if USE_TDM
        if (w == 0 && more) __builtin_amdgcn_s_wait_tensorcnt(0);
#endif
        __syncthreads();
    }

    // ---- epilogue ----
    #pragma unroll
    for (int mi = 0; mi < 2; ++mi)
        #pragma unroll
        for (int ni = 0; ni < 2; ++ni)
            #pragma unroll
            for (int e = 0; e < 8; ++e) {
                int m = bm + waveM * 32 + mi * 16 + e + hk * 8;
                int n = bn + waveN * 32 + ni * 16 + r;
                float v = acc[mi][ni][e];
                if (BIAS)    v += bias[n];
                if (OP == 1) v = 0.5f * v * (1.0f + erff(v * 0.70710678118f));
                if (OP == 2) v = (v > 0.0f) ? (v + 1.0f) : __expf(v);
                if (ROWSCALE) v *= rowscale[m];
                if (ADD_RES)  v += res[(size_t)m * N + n];
                if (STORE_F32) Cf[(size_t)m * N + n] = v;
                if (STORE_F16) Ch[(size_t)m * N + n] = (_Float16)v;
            }
}

// ---------------------------------------------------------------------
// kv = k'^T @ v  per batch:  [P=64, D=128], K-reduced over L with split-K
// grid = (L/64 splits, BATCH). Atomic f32 accumulation into kvf.
// Both operands staged transposed with packed l-pair ds_store_b32
// (lowers to ds_store_2addr_b32). 128 concurrent workgroups hide latency.
// ---------------------------------------------------------------------
__global__ __launch_bounds__(256)
void kv_splitk_kernel(const _Float16* __restrict__ kp, const _Float16* __restrict__ vv,
                      float* __restrict__ kvf)
{
    __shared__ _Float16 At[PROJ  * LDSTR];   // k'^T tile [p][l]
    __shared__ _Float16 Bt[DMODEL * LDSTR];  // v^T  tile [d][l]

    const int t    = threadIdx.x;
    const int lane = t & 31;
    const int w    = t >> 5;
    const int tm   = w & 3;     // which 16-row (P) tile
    const int nh   = w >> 2;    // which 64-col (D) half
    const int b    = blockIdx.y;
    const int l0   = blockIdx.x * 64;

    const _Float16* kpb = kp + (size_t)b * LSEQ * PROJ;
    const _Float16* vb  = vv + (size_t)b * LSEQ * DMODEL;

    v8f acc[4] = {};
    const int r  = lane & 15;
    const int hk = lane >> 4;

    for (int ks = 0; ks < 64; ks += 32) {
        // stage k'^T: 32 l-rows x 64 p -> At[p][l], packed l-pairs
        {
            int lp = t >> 4;   // 0..15 -> l = 2*lp
            int g  = t & 15;   // p-group of 4
            const _Float16* kr = kpb + (size_t)(l0 + ks + 2 * lp) * PROJ + g * 4;
            v4h r0 = *(const v4h*)kr;
            v4h r1 = *(const v4h*)(kr + PROJ);
            #pragma unroll
            for (int j = 0; j < 4; ++j) {
                v2h pr; pr[0] = r0[j]; pr[1] = r1[j];
                *(v2h*)(At + (g * 4 + j) * LDSTR + 2 * lp) = pr;
            }
        }
        // stage v^T: 32 l-rows x 128 d -> Bt[d][l], packed l-pairs
        {
            int lp = t >> 4;   // 0..15 -> l = 2*lp
            int dg = t & 15;   // d-group of 8
            const _Float16* vr = vb + (size_t)(l0 + ks + 2 * lp) * DMODEL + dg * 8;
            v8h r0 = *(const v8h*)vr;
            v8h r1 = *(const v8h*)(vr + DMODEL);
            #pragma unroll
            for (int j = 0; j < 8; ++j) {
                v2h pr; pr[0] = r0[j]; pr[1] = r1[j];
                *(v2h*)(Bt + (dg * 8 + j) * LDSTR + 2 * lp) = pr;
            }
        }
        __syncthreads();

        int row = tm * 16 + r;
        v8h lo = *(const v8h*)(At + row * LDSTR + hk * 16);
        v8h hi = *(const v8h*)(At + row * LDSTR + hk * 16 + 8);
        v16h afrag;
        #pragma unroll
        for (int i = 0; i < 8; ++i) { afrag[i] = lo[i]; afrag[8 + i] = hi[i]; }
        #pragma unroll
        for (int ni = 0; ni < 4; ++ni) {
            int c = nh * 64 + ni * 16 + r;
            v16h bfrag = *(const v16h*)(Bt + c * LDSTR + hk * 16);
            acc[ni] = __builtin_amdgcn_wmma_f32_16x16x32_f16(
                false, afrag, false, bfrag, (short)0, acc[ni], false, false);
        }
        __syncthreads();
    }

    #pragma unroll
    for (int ni = 0; ni < 4; ++ni)
        #pragma unroll
        for (int e = 0; e < 8; ++e) {
            int p = tm * 16 + e + hk * 8;
            int d = nh * 64 + ni * 16 + r;
            atomicAdd(&kvf[((size_t)b * PROJ + p) * DMODEL + d], acc[ni][e]);
        }
}

// ---------------------------------------------------------------------
// small helper kernels
// ---------------------------------------------------------------------
__global__ void cvt_f16_kernel(const float* __restrict__ s, _Float16* __restrict__ d, int n)
{
    int i = blockIdx.x * 256 + threadIdx.x;
    if (i < n) d[i] = (_Float16)s[i];
}

__global__ void zero_f32_kernel(float* __restrict__ p, int n)
{
    int i = blockIdx.x * 256 + threadIdx.x;
    if (i < n) p[i] = 0.0f;
}

__global__ void embed_kernel(const int* __restrict__ tok, const float* __restrict__ emb,
                             const float* __restrict__ pos,
                             float* __restrict__ h, _Float16* __restrict__ h16)
{
    int i = blockIdx.x * 256 + threadIdx.x;       // < MTOK * DMODEL
    int m = i >> 7, d = i & (DMODEL - 1);
    int l = m & (LSEQ - 1);
    float v = emb[(size_t)tok[m] * DMODEL + d] + pos[(size_t)l * DMODEL + d];
    h[i]   = v;
    h16[i] = (_Float16)v;
}

// z[m] = 1 / (sum_p q'[m,p] + eps)
__global__ void zrecip_kernel(const _Float16* __restrict__ qp, float* __restrict__ z)
{
    int m = blockIdx.x * 256 + threadIdx.x;
    const _Float16* row = qp + (size_t)m * PROJ;
    float s = 1e-8f;
    #pragma unroll 8
    for (int j = 0; j < PROJ; ++j) s += (float)row[j];
    z[m] = 1.0f / s;
}

// per-row LayerNorm of num (already *z) -> f16.  one wave per row, 4 elems/lane
__global__ __launch_bounds__(256)
void layernorm_kernel(const float* __restrict__ x, const float* __restrict__ g,
                      const float* __restrict__ b, _Float16* __restrict__ out)
{
    int w    = threadIdx.x >> 5;
    int lane = threadIdx.x & 31;
    int m    = blockIdx.x * 8 + w;
    const float* row = x + (size_t)m * DMODEL;

    float v[4], s = 0.0f;
    #pragma unroll
    for (int j = 0; j < 4; ++j) { v[j] = row[lane + j * 32]; s += v[j]; }
    #pragma unroll
    for (int o = 16; o > 0; o >>= 1) s += __shfl_xor(s, o, 32);
    float mu = s * (1.0f / DMODEL);
    float vs = 0.0f;
    #pragma unroll
    for (int j = 0; j < 4; ++j) { float d = v[j] - mu; vs += d * d; }
    #pragma unroll
    for (int o = 16; o > 0; o >>= 1) vs += __shfl_xor(vs, o, 32);
    float inv = rsqrtf(vs * (1.0f / DMODEL) + 1e-5f);
    #pragma unroll
    for (int j = 0; j < 4; ++j) {
        int d = lane + j * 32;
        out[(size_t)m * DMODEL + d] = (_Float16)((v[j] - mu) * inv * g[d] + b[d]);
    }
}

// ---------------------------------------------------------------------
// host-side orchestration
// ---------------------------------------------------------------------
extern "C" void kernel_launch(void* const* d_in, const int* in_sizes, int n_in,
                              void* d_out, int out_size, void* d_ws, size_t ws_size,
                              hipStream_t stream)
{
    (void)in_sizes; (void)n_in; (void)out_size; (void)ws_size;

    const int*   tokens = (const int*)  d_in[0];
    const float* embedT = (const float*)d_in[1];
    const float* posT   = (const float*)d_in[2];
    const float* Wq     = (const float*)d_in[3];
    const float* Wk     = (const float*)d_in[4];
    const float* Wv     = (const float*)d_in[5];
    const float* projW  = (const float*)d_in[6];
    const float* ln_g   = (const float*)d_in[7];
    const float* ln_b   = (const float*)d_in[8];
    const float* W1     = (const float*)d_in[9];
    const float* b1     = (const float*)d_in[10];
    const float* W2     = (const float*)d_in[11];
    const float* b2     = (const float*)d_in[12];
    const float* Wout   = (const float*)d_in[13];
    float* logits = (float*)d_out;

    // ---- workspace carve-up ----
    char* base = (char*)d_ws;
    size_t off = 0;
    auto alloc = [&](size_t bytes) -> char* {
        char* p = base + off;
        off += (bytes + 255) & ~(size_t)255;
        return p;
    };
    float*    h    = (float*)   alloc((size_t)MTOK * DMODEL * 4);
    _Float16* h16  = (_Float16*)alloc((size_t)MTOK * DMODEL * 2);
    _Float16* q16  = (_Float16*)alloc((size_t)MTOK * DMODEL * 2);
    _Float16* k16  = (_Float16*)alloc((size_t)MTOK * DMODEL * 2);
    _Float16* v16  = (_Float16*)alloc((size_t)MTOK * DMODEL * 2);
    _Float16* qp16 = (_Float16*)alloc((size_t)MTOK * PROJ * 2);
    _Float16* kp16 = (_Float16*)alloc((size_t)MTOK * PROJ * 2);
    float*    z    = (float*)   alloc((size_t)MTOK * 4);
    float*    kvf  = (float*)   alloc((size_t)BATCH * PROJ * DMODEL * 4);
    _Float16* kv16 = (_Float16*)alloc((size_t)BATCH * PROJ * DMODEL * 2);
    float*    numf = (float*)   alloc((size_t)MTOK * DMODEL * 4);
    _Float16* a16  = (_Float16*)alloc((size_t)MTOK * DMODEL * 2);
    _Float16* f116 = (_Float16*)alloc((size_t)MTOK * DFF * 2);
    _Float16* Wq16   = (_Float16*)alloc((size_t)NLAYERS * DMODEL * DMODEL * 2);
    _Float16* Wk16   = (_Float16*)alloc((size_t)NLAYERS * DMODEL * DMODEL * 2);
    _Float16* Wv16   = (_Float16*)alloc((size_t)NLAYERS * DMODEL * DMODEL * 2);
    _Float16* proj16 = (_Float16*)alloc((size_t)NLAYERS * DMODEL * PROJ * 2);
    _Float16* W116   = (_Float16*)alloc((size_t)NLAYERS * DMODEL * DFF * 2);
    _Float16* W216   = (_Float16*)alloc((size_t)NLAYERS * DFF * DMODEL * 2);
    _Float16* Wout16 = (_Float16*)alloc((size_t)DMODEL * VOCAB * 2);

    auto cvt = [&](const float* s, _Float16* d, int n) {
        cvt_f16_kernel<<<(n + 255) / 256, 256, 0, stream>>>(s, d, n);
    };

    // ---- weight conversion to f16 (L2-resident after first touch) ----
    cvt(Wq,    Wq16,   NLAYERS * DMODEL * DMODEL);
    cvt(Wk,    Wk16,   NLAYERS * DMODEL * DMODEL);
    cvt(Wv,    Wv16,   NLAYERS * DMODEL * DMODEL);
    cvt(projW, proj16, NLAYERS * DMODEL * PROJ);
    cvt(W1,    W116,   NLAYERS * DMODEL * DFF);
    cvt(W2,    W216,   NLAYERS * DFF * DMODEL);
    cvt(Wout,  Wout16, DMODEL * VOCAB);

    // ---- embedding + positional ----
    embed_kernel<<<(MTOK * DMODEL) / 256, 256, 0, stream>>>(tokens, embedT, posT, h, h16);

    for (int i = 0; i < NLAYERS; ++i) {
        const _Float16* wq = Wq16   + (size_t)i * DMODEL * DMODEL;
        const _Float16* wk = Wk16   + (size_t)i * DMODEL * DMODEL;
        const _Float16* wv = Wv16   + (size_t)i * DMODEL * DMODEL;
        const _Float16* pr = proj16 + (size_t)i * DMODEL * PROJ;
        const _Float16* w1 = W116   + (size_t)i * DMODEL * DFF;
        const _Float16* w2 = W216   + (size_t)i * DFF * DMODEL;
        const float* bias1 = b1 + (size_t)i * DFF;
        const float* bias2 = b2 + (size_t)i * DMODEL;
        const float* g     = ln_g + (size_t)i * DMODEL;
        const float* bb    = ln_b + (size_t)i * DMODEL;

        dim3 gQKV(DMODEL / 64, MTOK / 128);
        // q = h @ Wq ; k = h @ Wk ; v = h @ Wv   (f16 outputs)
        gemm_f16_kernel<0, false, false, true, false, false><<<gQKV, 256, 0, stream>>>(
            h16, wq, nullptr, nullptr, nullptr, nullptr, q16, MTOK, DMODEL, DMODEL);
        gemm_f16_kernel<0, false, false, true, false, false><<<gQKV, 256, 0, stream>>>(
            h16, wk, nullptr, nullptr, nullptr, nullptr, k16, MTOK, DMODEL, DMODEL);
        gemm_f16_kernel<0, false, false, true, false, false><<<gQKV, 256, 0, stream>>>(
            h16, wv, nullptr, nullptr, nullptr, nullptr, v16, MTOK, DMODEL, DMODEL);

        // q' = elu(q@proj)+1 ; k' = elu(k@proj)+1
        dim3 gP(PROJ / 64, MTOK / 128);
        gemm_f16_kernel<2, false, false, true, false, false><<<gP, 256, 0, stream>>>(
            q16, pr, nullptr, nullptr, nullptr, nullptr, qp16, MTOK, PROJ, DMODEL);
        gemm_f16_kernel<2, false, false, true, false, false><<<gP, 256, 0, stream>>>(
            k16, pr, nullptr, nullptr, nullptr, nullptr, kp16, MTOK, PROJ, DMODEL);

        // kv = k'^T @ v   (split-K atomic f32, then -> f16)
        zero_f32_kernel<<<(BATCH * PROJ * DMODEL) / 256, 256, 0, stream>>>(
            kvf, BATCH * PROJ * DMODEL);
        kv_splitk_kernel<<<dim3(LSEQ / 64, BATCH), 256, 0, stream>>>(kp16, v16, kvf);
        cvt(kvf, kv16, BATCH * PROJ * DMODEL);

        // z = 1/(rowsum(q') + eps)
        zrecip_kernel<<<MTOK / 256, 256, 0, stream>>>(qp16, z);

        // num = q' @ kv, scaled by z  (per batch, kv differs)
        for (int b = 0; b < BATCH; ++b) {
            dim3 gN(DMODEL / 64, LSEQ / 128);
            gemm_f16_kernel<0, false, true, false, false, true><<<gN, 256, 0, stream>>>(
                qp16 + (size_t)b * LSEQ * PROJ,
                kv16 + (size_t)b * PROJ * DMODEL,
                nullptr, nullptr,
                z + (size_t)b * LSEQ,
                numf + (size_t)b * LSEQ * DMODEL, nullptr,
                LSEQ, DMODEL, PROJ);
        }

        // a = LayerNorm(attn)
        layernorm_kernel<<<MTOK / 8, 256, 0, stream>>>(numf, g, bb, a16);

        // f1 = gelu(a @ W1 + b1)
        dim3 gF1(DFF / 64, MTOK / 128);
        gemm_f16_kernel<1, true, false, true, false, false><<<gF1, 256, 0, stream>>>(
            a16, w1, bias1, nullptr, nullptr, nullptr, f116, MTOK, DFF, DMODEL);

        // h = h + f1 @ W2 + b2   (writes new residual f32 + f16 copy)
        dim3 gF2(DMODEL / 64, MTOK / 128);
        gemm_f16_kernel<0, true, true, true, true, false><<<gF2, 256, 0, stream>>>(
            f116, w2, bias2, h, nullptr, h, h16, MTOK, DMODEL, DFF);
    }

    // logits = h @ Wout  (f32 straight to d_out)
    dim3 gL(VOCAB / 64, MTOK / 128);
    gemm_f16_kernel<0, false, true, false, false, false><<<gL, 256, 0, stream>>>(
        h16, Wout16, nullptr, nullptr, nullptr, logits, nullptr, MTOK, VOCAB, DMODEL);
}
